// SequentialHyperMamba_60713657697149
// MI455X (gfx1250) — matrix operand order repaired and checked by reading, not compiled, not run on hardware
//
#include <hip/hip_runtime.h>
#include <hip/hip_bf16.h>
#include <math.h>

// ---------------- problem constants ----------------
#define BB   4
#define CCH  256
#define HWsz 4096          // 64*64
#define LL   HWsz
#define MM   (BB*LL)       // 16384 token rows
#define EE   8
#define NSt  8
#define RR   16
#define NCH  32            // scan chunks per sequence
#define CHL  (LL/NCH)      // 128 steps per chunk

typedef __bf16 bf16_t;
typedef __attribute__((ext_vector_type(16))) __bf16 v16bf;
typedef __attribute__((ext_vector_type(8)))  float  v8f;

struct __align__(16) bf8chunk { __bf16 v[8]; };   // one b128 load

__device__ __forceinline__ float silu_f(float x){ return x / (1.f + __expf(-x)); }
__device__ __forceinline__ float softplus_f(float x){
  return (x > 20.f) ? x : log1pf(__expf(x));
}

// ---------------- tiny utility kernels ----------------
__global__ void k_zero(float* p, int n){
  int i = blockIdx.x*blockDim.x + threadIdx.x;
  if (i < n) p[i] = 0.f;
}

// src f32 [K,N] row-major -> dst bf16 [N,Kpad] row-major (zero-padded K)
__global__ void k_wt_transpose(const float* __restrict__ src, bf16_t* __restrict__ dst,
                               int K, int N, int Kpad){
  int i = blockIdx.x*blockDim.x + threadIdx.x;
  if (i >= N*Kpad) return;
  int n = i / Kpad, k = i % Kpad;
  dst[i] = (k < K) ? (bf16_t)src[k*N + n] : (bf16_t)(0.0f);
}

__global__ void k_wt_convert(const float* __restrict__ src, bf16_t* __restrict__ dst, int n){
  int i = blockIdx.x*blockDim.x + threadIdx.x;
  if (i < n) dst[i] = (bf16_t)src[i];
}

// ---------------- HyperACE kernels ----------------
__global__ void k_hyper_A(const float* __restrict__ x, const float* __restrict__ W_edge,
                          float* __restrict__ Apart, float* __restrict__ sumA){
  __shared__ float wsh[CCH*EE];
  int tid = threadIdx.x;
  for (int i = tid; i < CCH*EE; i += blockDim.x) wsh[i] = W_edge[i];
  __syncthreads();
  int gid = blockIdx.x*blockDim.x + tid;     // [0, B*HW)
  int b = gid / HWsz, n = gid % HWsz;
  float lg[EE];
  #pragma unroll
  for (int e=0;e<EE;e++) lg[e] = 0.f;
  for (int c=0;c<CCH;c++){
    float xv = x[((size_t)b*CCH + c)*HWsz + n];   // coalesced across threads (n)
    #pragma unroll
    for (int e=0;e<EE;e++) lg[e] += xv * wsh[c*EE+e];
  }
  float mx = lg[0];
  #pragma unroll
  for (int e=1;e<EE;e++) mx = fmaxf(mx, lg[e]);
  float s = 0.f;
  #pragma unroll
  for (int e=0;e<EE;e++){ lg[e] = __expf(lg[e]-mx); s += lg[e]; }
  float inv = 1.f/s;
  #pragma unroll
  for (int e=0;e<EE;e++){
    float a = lg[e]*inv;
    Apart[(size_t)gid*EE + e] = a;
    atomicAdd(&sumA[b*EE + e], a);
  }
}

__global__ void k_edge_reduce(const float* __restrict__ x, const float* __restrict__ Apart,
                              const float* __restrict__ sumA, float* __restrict__ edgeN){
  int be = blockIdx.x;
  int b = be / EE, e = be % EE;
  int c = threadIdx.x;
  const float* xrow = x + ((size_t)b*CCH + c)*HWsz;
  const float* arow = Apart + (size_t)b*HWsz*EE + e;
  float acc = 0.f;
  for (int n=0;n<HWsz;n++) acc += arow[(size_t)n*EE] * xrow[n];
  edgeN[(size_t)be*CCH + c] = acc / (sumA[be] + 1e-6f);
}

__global__ void k_edge_mlp(const float* __restrict__ edgeN, const float* __restrict__ W_e,
                           const float* __restrict__ b_e, float* __restrict__ edge2){
  __shared__ float sh[CCH];
  int be = blockIdx.x; int o = threadIdx.x;
  sh[o] = edgeN[(size_t)be*CCH + o];
  __syncthreads();
  float acc = b_e[o];
  for (int c=0;c<CCH;c++) acc += sh[c]*W_e[(size_t)c*CCH + o];
  edge2[(size_t)be*CCH + o] = silu_f(acc);
}

__global__ void k_pre_fuse(const float* __restrict__ x, const float* __restrict__ Apart,
                           const float* __restrict__ edge2, bf16_t* __restrict__ pre){
  __shared__ float ash[EE];
  int m = blockIdx.x; int c = threadIdx.x;
  int b = m / HWsz, n = m % HWsz;
  if (c < EE) ash[c] = Apart[(size_t)m*EE + c];
  __syncthreads();
  float acc = x[((size_t)b*CCH + c)*HWsz + n];
  #pragma unroll
  for (int e=0;e<EE;e++) acc += ash[e] * edge2[(size_t)(b*EE+e)*CCH + c];
  pre[(size_t)m*CCH + c] = (bf16_t)acc;
}

// ---------------- WMMA GEMM: C[M,N] = A[M,K] @ B[K,N] ----------------
// A bf16 [M,K] row-major, Bt bf16 [N,K] row-major (pre-transposed weights).
// One wave32 per block computes a 16 x (NT*16) tile: one A fragment is
// reused for NT WMMA ops per K-step (4x less A traffic for NT=4).
// EPI: 0=plain f32 store  1=bias+SiLU f32 store  2=bias+SiLU scatter to [B,OC,HW]
template<int EPI, int NT>
__global__ void k_wmma_gemm(const bf16_t* __restrict__ A, const bf16_t* __restrict__ Bt,
                            const float* __restrict__ bias, float* __restrict__ out,
                            int Ndim, int Kdim, int HWp, int OCp){
  const int lane = threadIdx.x;
  const int half = lane >> 4;        // lane group 0/1
  const int l16  = lane & 15;
  const int tm = blockIdx.y;
  const int ncol0 = blockIdx.x * (NT*16);
  const bf16_t* Abase = A + (size_t)(tm*16 + l16)*Kdim;   // row M = l16 in both halves
  const bf16_t* Bbase[NT];
  #pragma unroll
  for (int t=0;t<NT;t++) Bbase[t] = Bt + (size_t)(ncol0 + t*16 + l16)*Kdim;
  v8f acc[NT] = {};
  for (int k0 = 0; k0 < Kdim; k0 += 32){
    // A layout: lanes<16: K {k0..k0+7, k0+16..k0+23}; lanes>=16: +8
    int ka = k0 + (half ? 8 : 0);
    bf8chunk a0 = *(const bf8chunk*)(Abase + ka);
    bf8chunk a1 = *(const bf8chunk*)(Abase + ka + 16);
    v16bf av;
    #pragma unroll
    for (int i=0;i<8;i++){ av[i] = a0.v[i]; av[i+8] = a1.v[i]; }
    // B layout: lanes<16: K {k0..k0+7, k0+16..k0+23}; lanes>=16: +8
    int kb = k0 + half*8;
    #pragma unroll
    for (int t=0;t<NT;t++){
      bf8chunk b0 = *(const bf8chunk*)(Bbase[t] + kb);
      bf8chunk b1 = *(const bf8chunk*)(Bbase[t] + kb + 16);
      v16bf bv;
      #pragma unroll
      for (int i=0;i<8;i++){ bv[i] = b0.v[i]; bv[i+8] = b1.v[i]; }
      acc[t] = __builtin_amdgcn_wmma_f32_16x16x32_bf16(
          /*neg_a=*/false, av, /*neg_b=*/false, bv,
          /*c_mod=*/(short)0, acc[t], /*reuse_a=*/false, /*reuse_b=*/false);
    }
  }
  #pragma unroll
  for (int t=0;t<NT;t++){
    int col = ncol0 + t*16 + l16;
    #pragma unroll
    for (int r=0;r<8;r++){
      int m = tm*16 + half*8 + r;      // D: VGPR r -> row r (+8 for lanes>=16)
      float v = acc[t][r];
      if (EPI == 0){
        out[(size_t)m*Ndim + col] = v;
      } else if (EPI == 1){
        v += bias[col];
        out[(size_t)m*Ndim + col] = silu_f(v);
      } else {
        v += bias[col];
        int b = m / HWp, n = m % HWp;
        out[((size_t)b*OCp + col)*HWp + n] = silu_f(v);
      }
    }
  }
}

// ---------------- RMSNorm -> bf16 (per token row over C=256) ----------------
__global__ void k_rmsnorm_bf16(const float* __restrict__ src, const float* __restrict__ w,
                               bf16_t* __restrict__ dst){
  __shared__ float red[CCH];
  int m = blockIdx.x, c = threadIdx.x;
  float v = src[(size_t)m*CCH + c];
  red[c] = v*v;
  __syncthreads();
  for (int s = CCH/2; s > 0; s >>= 1){
    if (c < s) red[c] += red[c+s];
    __syncthreads();
  }
  float nrm = sqrtf(red[0]);                 // L2 norm
  float scale = 1.f/(nrm/16.f + 1e-6f);      // sqrt(C)=16
  dst[(size_t)m*CCH + c] = (bf16_t)(v*scale*w[c]);
}

// ---------------- depthwise causal conv (K=4) + SiLU, optional flip ----------------
__global__ void k_conv_silu(const float* __restrict__ xz, const float* __restrict__ cw,
                            const float* __restrict__ cb, float* __restrict__ xcf,
                            bf16_t* __restrict__ xcb, int flip){
  int m = blockIdx.x, c = threadIdx.x;
  int b = m / LL, l = m % LL;
  float acc = cb[c];
  #pragma unroll
  for (int k=0;k<4;k++){
    int j = l - 3 + k;
    if (j >= 0){
      int js = flip ? (LL-1-j) : j;
      acc += cw[c*4+k] * xz[((size_t)b*LL + js)*(2*CCH) + c];
    }
  }
  float v = silu_f(acc);
  xcf[(size_t)m*CCH + c] = v;
  xcb[(size_t)m*CCH + c] = (bf16_t)v;
}

// dt_bf16[m,k] : cols 0..15 from dbl, 16..31 zero (pads K to 32 for WMMA)
__global__ void k_dt_prep(const float* __restrict__ dbl, bf16_t* __restrict__ dt){
  int i = blockIdx.x*blockDim.x + threadIdx.x;   // M*32
  int k = i & 31;
  dt[i] = (k < RR) ? (bf16_t)dbl[i] : (bf16_t)(0.0f);
}

// ---------------- chunked selective scan ----------------
// Recurrence h <- dA*h + dBu is affine (diagonal) per (b,c,n); decompose into
// NCH chunks: pass1 computes per-chunk (P = prod dA, q = h_end | h0=0);
// mid combines chunk states serially (32 steps, trivial); pass2 replays each
// chunk from its correct h0 and emits y. Serial depth 4096 -> 2*128, waves 32 -> 1024.

__device__ __forceinline__ void load_f16vals(const float* __restrict__ p, float* dst, int cnt4){
  const float4* p4 = (const float4*)p;
  #pragma unroll
  for (int q=0;q<4;q++){
    if (q < cnt4){
      float4 t = p4[q];
      dst[4*q+0]=t.x; dst[4*q+1]=t.y; dst[4*q+2]=t.z; dst[4*q+3]=t.w;
    }
  }
}

__global__ void k_scan_pass1(const float* __restrict__ dtp, const float* __restrict__ xc,
                             const float* __restrict__ dbl, const float* __restrict__ bdt,
                             const float* __restrict__ Alog,
                             float* __restrict__ chP, float* __restrict__ chH){
  int bcix = blockIdx.x;               // b*NCH + ch
  int b = bcix / NCH, ch = bcix % NCH;
  int c = threadIdx.x;
  float Aneg[NSt];
  #pragma unroll
  for (int n=0;n<NSt;n++) Aneg[n] = -__expf(Alog[c*NSt + n]);
  float h[NSt] = {}, P[NSt];
  #pragma unroll
  for (int n=0;n<NSt;n++) P[n] = 1.f;
  float bd = bdt[c];
  for (int l = ch*CHL; l < (ch+1)*CHL; ++l){
    size_t m = (size_t)b*LL + l;
    float bm[8];
    load_f16vals(dbl + m*32 + 16, bm, 2);      // Bm only
    float delta = softplus_f(dtp[m*CCH + c] + bd);
    float du = delta * xc[m*CCH + c];
    #pragma unroll
    for (int n=0;n<NSt;n++){
      float dA = __expf(delta*Aneg[n]);
      h[n] = h[n]*dA + du*bm[n];
      P[n] *= dA;
    }
  }
  size_t o = ((size_t)bcix*CCH + c)*NSt;
  #pragma unroll
  for (int n=0;n<NSt;n++){ chP[o+n] = P[n]; chH[o+n] = h[n]; }
}

__global__ void k_scan_mid(const float* __restrict__ chP, const float* __restrict__ chH,
                           float* __restrict__ chH0){
  int b = blockIdx.x, c = threadIdx.x;
  float h0[NSt] = {};
  for (int ch=0; ch<NCH; ++ch){
    size_t o = (((size_t)b*NCH + ch)*CCH + c)*NSt;
    #pragma unroll
    for (int n=0;n<NSt;n++) chH0[o+n] = h0[n];
    #pragma unroll
    for (int n=0;n<NSt;n++) h0[n] = chP[o+n]*h0[n] + chH[o+n];
  }
}

__global__ void k_scan_pass2(const float* __restrict__ dtp, const float* __restrict__ xc,
                             const float* __restrict__ dbl, const float* __restrict__ bdt,
                             const float* __restrict__ Alog, const float* __restrict__ Dp,
                             const float* __restrict__ chH0, float* __restrict__ y){
  int bcix = blockIdx.x;
  int b = bcix / NCH, ch = bcix % NCH;
  int c = threadIdx.x;
  float Aneg[NSt];
  #pragma unroll
  for (int n=0;n<NSt;n++) Aneg[n] = -__expf(Alog[c*NSt + n]);
  float h[NSt];
  {
    size_t o = ((size_t)bcix*CCH + c)*NSt;
    #pragma unroll
    for (int n=0;n<NSt;n++) h[n] = chH0[o+n];
  }
  float Dc = Dp[c], bd = bdt[c];
  for (int l = ch*CHL; l < (ch+1)*CHL; ++l){
    size_t m = (size_t)b*LL + l;
    float bcv[16];
    load_f16vals(dbl + m*32 + 16, bcv, 4);     // Bm[0..7], Cm[8..15]
    float delta = softplus_f(dtp[m*CCH + c] + bd);
    float u = xc[m*CCH + c];
    float du = delta*u;
    float yv = u*Dc;
    #pragma unroll
    for (int n=0;n<NSt;n++){
      h[n] = h[n]*__expf(delta*Aneg[n]) + du*bcv[n];
      yv  += h[n]*bcv[8+n];
    }
    y[m*CCH + c] = yv;
  }
}

// ycomb = 0.5*(yf + flip(yb)) * silu(z)
__global__ void k_combine(const float* __restrict__ yf, const float* __restrict__ yb,
                          const float* __restrict__ xz, float* __restrict__ ycomb){
  int m = blockIdx.x, c = threadIdx.x;
  int b = m / LL, l = m % LL;
  float z = xz[(size_t)m*(2*CCH) + CCH + c];
  float v = 0.5f*(yf[(size_t)m*CCH + c] + yb[((size_t)b*LL + (LL-1-l))*CCH + c]) * silu_f(z);
  ycomb[(size_t)m*CCH + c] = v;
}

// final residual: x + 2*x_h_seq + y2*gamma  (bf16 A-matrix for W_op GEMM)
__global__ void k_fsum(const float* __restrict__ x, const float* __restrict__ xhseq,
                       const float* __restrict__ y2, const float* __restrict__ gamma,
                       bf16_t* __restrict__ fs){
  int m = blockIdx.x, c = threadIdx.x;
  int b = m / HWsz, n = m % HWsz;
  float v = x[((size_t)b*CCH + c)*HWsz + n]
          + 2.f*xhseq[(size_t)m*CCH + c]
          + y2[(size_t)m*CCH + c]*gamma[c];
  fs[(size_t)m*CCH + c] = (bf16_t)v;
}

// ---------------- launch ----------------
extern "C" void kernel_launch(void* const* d_in, const int* in_sizes, int n_in,
                              void* d_out, int out_size, void* d_ws, size_t ws_size,
                              hipStream_t stream) {
  const float* x       = (const float*)d_in[0];
  const float* W_edge  = (const float*)d_in[1];
  const float* W_e     = (const float*)d_in[2];
  const float* b_e     = (const float*)d_in[3];
  const float* W_fuse  = (const float*)d_in[4];
  const float* b_fuse  = (const float*)d_in[5];
  const float* norm_w  = (const float*)d_in[6];
  const float* gamma   = (const float*)d_in[7];
  const float* W_in    = (const float*)d_in[8];
  const float* conv_w_f= (const float*)d_in[9];
  const float* conv_b_f= (const float*)d_in[10];
  const float* Wx_f    = (const float*)d_in[11];
  const float* Wdt_f   = (const float*)d_in[12];
  const float* bdt_f   = (const float*)d_in[13];
  const float* Alog_f  = (const float*)d_in[14];
  const float* D_f     = (const float*)d_in[15];
  const float* conv_w_b= (const float*)d_in[16];
  const float* conv_b_b= (const float*)d_in[17];
  const float* Wx_b    = (const float*)d_in[18];
  const float* Wdt_b   = (const float*)d_in[19];
  const float* bdt_b   = (const float*)d_in[20];
  const float* Alog_b  = (const float*)d_in[21];
  const float* D_b     = (const float*)d_in[22];
  const float* mnorm_w = (const float*)d_in[23];
  const float* W_out   = (const float*)d_in[24];
  const float* W_op    = (const float*)d_in[25];
  const float* b_op    = (const float*)d_in[26];
  float* out = (float*)d_out;

  // ---- workspace arena (aliased lifetimes) ----
  uint8_t* ws = (uint8_t*)d_ws;
  size_t off = 0;
  auto take = [&](size_t bytes)->void*{
    void* p = ws + off;
    off += (bytes + 255) & ~(size_t)255;
    return p;
  };
  float*  Apart  = (float*) take((size_t)MM*EE*4);
  float*  sumA   = (float*) take(BB*EE*4);
  float*  edgeN  = (float*) take((size_t)BB*EE*CCH*4);
  float*  edge2  = (float*) take((size_t)BB*EE*CCH*4);
  bf16_t* prefuse= (bf16_t*)take((size_t)MM*CCH*2);
  float*  xhseq  = (float*) take((size_t)MM*CCH*4);
  bf16_t* sbuf   = (bf16_t*)take((size_t)MM*CCH*2);   // s, later s2
  float*  xz     = (float*) take((size_t)MM*2*CCH*4); // [M,512]: xm | z
  float*  xcf    = (float*) take((size_t)MM*CCH*4);   // xc f32; later y2
  bf16_t* xcb    = (bf16_t*)take((size_t)MM*CCH*2);   // xc bf16; later fsum
  float*  dbl    = (float*) take((size_t)MM*32*4);
  bf16_t* dtbf   = (bf16_t*)take((size_t)MM*32*2);
  float*  dtp    = (float*) take((size_t)MM*CCH*4);   // delta-proj; later ycomb
  float*  yf     = (float*) take((size_t)MM*CCH*4);
  float*  yb     = (float*) take((size_t)MM*CCH*4);
  float*  chP    = (float*) take((size_t)BB*NCH*CCH*NSt*4);
  float*  chH    = (float*) take((size_t)BB*NCH*CCH*NSt*4);
  float*  chH0   = (float*) take((size_t)BB*NCH*CCH*NSt*4);
  bf16_t* Wfuse_t= (bf16_t*)take((size_t)CCH*CCH*2);
  bf16_t* Win_t  = (bf16_t*)take((size_t)2*CCH*CCH*2);
  bf16_t* Wxf_t  = (bf16_t*)take((size_t)32*CCH*2);
  bf16_t* Wxb_t  = (bf16_t*)take((size_t)32*CCH*2);
  bf16_t* Wdtf_t = (bf16_t*)take((size_t)CCH*32*2);
  bf16_t* Wdtb_t = (bf16_t*)take((size_t)CCH*32*2);
  bf16_t* Wout_t = (bf16_t*)take((size_t)CCH*CCH*2);
  bf16_t* Wop_b  = (bf16_t*)take((size_t)CCH*CCH*2);

  // ---- weight prep (bf16 transpose / convert) ----
  k_zero<<<1, 32, 0, stream>>>(sumA, BB*EE);
  k_wt_transpose<<<(CCH*CCH+255)/256, 256, 0, stream>>>(W_fuse, Wfuse_t, CCH, CCH, CCH);
  k_wt_transpose<<<(2*CCH*CCH+255)/256, 256, 0, stream>>>(W_in, Win_t, CCH, 2*CCH, CCH);
  k_wt_transpose<<<(32*CCH+255)/256, 256, 0, stream>>>(Wx_f, Wxf_t, CCH, 32, CCH);
  k_wt_transpose<<<(32*CCH+255)/256, 256, 0, stream>>>(Wx_b, Wxb_t, CCH, 32, CCH);
  k_wt_transpose<<<(CCH*32+255)/256, 256, 0, stream>>>(Wdt_f, Wdtf_t, RR, CCH, 32);
  k_wt_transpose<<<(CCH*32+255)/256, 256, 0, stream>>>(Wdt_b, Wdtb_t, RR, CCH, 32);
  k_wt_transpose<<<(CCH*CCH+255)/256, 256, 0, stream>>>(W_out, Wout_t, CCH, CCH, CCH);
  k_wt_convert<<<(CCH*CCH+255)/256, 256, 0, stream>>>(W_op, Wop_b, CCH*CCH);

  // ---- HyperACE ----
  k_hyper_A<<<MM/256, 256, 0, stream>>>(x, W_edge, Apart, sumA);
  k_edge_reduce<<<BB*EE, CCH, 0, stream>>>(x, Apart, sumA, edgeN);
  k_edge_mlp<<<BB*EE, CCH, 0, stream>>>(edgeN, W_e, b_e, edge2);
  k_pre_fuse<<<MM, CCH, 0, stream>>>(x, Apart, edge2, prefuse);

  // x_h_seq = silu(pre_fuse @ W_fuse + b_fuse)   [WMMA]
  k_wmma_gemm<1,4><<<dim3(CCH/64, MM/16), 32, 0, stream>>>(prefuse, Wfuse_t, b_fuse, xhseq, CCH, CCH, 0, 0);

  // ---- Mamba front: s = rmsnorm(seq); xz = s @ W_in   [WMMA] ----
  k_rmsnorm_bf16<<<MM, CCH, 0, stream>>>(xhseq, norm_w, sbuf);
  k_wmma_gemm<0,4><<<dim3(2*CCH/64, MM/16), 32, 0, stream>>>(sbuf, Win_t, nullptr, xz, 2*CCH, CCH, 0, 0);

  // ---- forward direction ----
  k_conv_silu<<<MM, CCH, 0, stream>>>(xz, conv_w_f, conv_b_f, xcf, xcb, 0);
  k_wmma_gemm<0,2><<<dim3(1, MM/16), 32, 0, stream>>>(xcb, Wxf_t, nullptr, dbl, 32, CCH, 0, 0);
  k_dt_prep<<<MM*32/256, 256, 0, stream>>>(dbl, dtbf);
  k_wmma_gemm<0,4><<<dim3(CCH/64, MM/16), 32, 0, stream>>>(dtbf, Wdtf_t, nullptr, dtp, CCH, 32, 0, 0);
  k_scan_pass1<<<BB*NCH, CCH, 0, stream>>>(dtp, xcf, dbl, bdt_f, Alog_f, chP, chH);
  k_scan_mid<<<BB, CCH, 0, stream>>>(chP, chH, chH0);
  k_scan_pass2<<<BB*NCH, CCH, 0, stream>>>(dtp, xcf, dbl, bdt_f, Alog_f, D_f, chH0, yf);

  // ---- backward direction (stored flipped; unflipped at combine) ----
  k_conv_silu<<<MM, CCH, 0, stream>>>(xz, conv_w_b, conv_b_b, xcf, xcb, 1);
  k_wmma_gemm<0,2><<<dim3(1, MM/16), 32, 0, stream>>>(xcb, Wxb_t, nullptr, dbl, 32, CCH, 0, 0);
  k_dt_prep<<<MM*32/256, 256, 0, stream>>>(dbl, dtbf);
  k_wmma_gemm<0,4><<<dim3(CCH/64, MM/16), 32, 0, stream>>>(dtbf, Wdtb_t, nullptr, dtp, CCH, 32, 0, 0);
  k_scan_pass1<<<BB*NCH, CCH, 0, stream>>>(dtp, xcf, dbl, bdt_b, Alog_b, chP, chH);
  k_scan_mid<<<BB, CCH, 0, stream>>>(chP, chH, chH0);
  k_scan_pass2<<<BB*NCH, CCH, 0, stream>>>(dtp, xcf, dbl, bdt_b, Alog_b, D_b, chH0, yb);

  // ---- gate + out-proj: y2 = rmsnorm(ycomb, mnorm) @ W_out   [WMMA] ----
  k_combine<<<MM, CCH, 0, stream>>>(yf, yb, xz, dtp /*=ycomb*/);
  k_rmsnorm_bf16<<<MM, CCH, 0, stream>>>(dtp, mnorm_w, sbuf /*=s2*/);
  k_wmma_gemm<0,4><<<dim3(CCH/64, MM/16), 32, 0, stream>>>(sbuf, Wout_t, nullptr, xcf /*=y2*/, CCH, CCH, 0, 0);

  // ---- fused residual + 1x1 out conv + SiLU, scattered to [B,C,H,W]   [WMMA] ----
  k_fsum<<<MM, CCH, 0, stream>>>(x, xhseq, xcf, gamma, xcb /*=fsum bf16*/);
  k_wmma_gemm<2,4><<<dim3(CCH/64, MM/16), 32, 0, stream>>>(xcb, Wop_b, b_op, out, CCH, CCH, HWsz, CCH);

  (void)in_sizes; (void)n_in; (void)out_size; (void)ws_size;
}
